// MyGenerator_84482006713322
// MI455X (gfx1250) — matrix operand order repaired and checked by reading, not compiled, not run on hardware
//
#include <hip/hip_runtime.h>

typedef __attribute__((ext_vector_type(16))) _Float16 v16h;
typedef __attribute__((ext_vector_type(8)))  float    v8f;
typedef __attribute__((ext_vector_type(4)))  float    v4f;

// ---------------- problem sizes ----------------
// B=2, NCH=8 classes (process 1..7 -> 7), H=256, HS=64, C=256
#define HW      4096      // 64*64
#define CDIM    256
#define NCOMBO  14        // 2 batches * 7 classes

// ---------------- workspace layout ----------------
static constexpr size_t alignup(size_t x){ return (x + 255) & ~(size_t)255; }
static constexpr size_t OFF_TMASK = 0;                                          // [14][4096] f32
static constexpr size_t OFF_SMASK = alignup(OFF_TMASK + (size_t)NCOMBO*HW*4);   // [14][4096] f32
static constexpr size_t OFF_CNTS  = alignup(OFF_SMASK + (size_t)NCOMBO*HW*4);   // [32] f32: t_cnt[14], s_cnt[14]
static constexpr size_t OFF_TCOL  = alignup(OFF_CNTS  + 32*4);                  // [2][3][4096] f32
static constexpr size_t OFF_SCOL  = alignup(OFF_TCOL  + (size_t)2*3*HW*4);
static constexpr size_t OFF_MEANQ = alignup(OFF_SCOL  + (size_t)2*3*HW*4);      // [14][256] f32
static constexpr size_t OFF_MEANS = alignup(OFF_MEANQ + (size_t)NCOMBO*CDIM*4);
static constexpr size_t OFF_QH    = alignup(OFF_MEANS + (size_t)NCOMBO*CDIM*4); // [14][4096][256] f16
static constexpr size_t OFF_SH    = alignup(OFF_QH + (size_t)NCOMBO*HW*CDIM*2);
static constexpr size_t OFF_CANV  = alignup(OFF_SH + (size_t)NCOMBO*HW*CDIM*2); // [2][3][4096] f32

// ---------------- kernel 0: masks (nearest-down) + counts ----------------
__global__ void k_masks(const float* __restrict__ t_oh, const float* __restrict__ s_oh,
                        float* __restrict__ tmask, float* __restrict__ smask,
                        float* __restrict__ cnts) {
    int combo = blockIdx.x;            // 0..13
    int b = combo / 7, k = combo % 7 + 1;
    const float* tsrc = t_oh + (size_t)(b*8 + k) * 65536;
    const float* ssrc = s_oh + (size_t)(b*8 + k) * 65536;
    float tacc = 0.f, sacc = 0.f;
    for (int i = threadIdx.x; i < HW; i += 256) {
        int src = (i >> 6) * 1024 + (i & 63) * 4;     // (4y)*256 + 4x
        float tv = tsrc[src] > 0.f ? 1.f : 0.f;
        float sv = ssrc[src] > 0.f ? 1.f : 0.f;
        tmask[(size_t)combo*HW + i] = tv;
        smask[(size_t)combo*HW + i] = sv;
        tacc += tv; sacc += sv;
    }
    __shared__ float st[256], ss[256];
    st[threadIdx.x] = tacc; ss[threadIdx.x] = sacc; __syncthreads();
    for (int s = 128; s > 0; s >>= 1) {
        if (threadIdx.x < s) { st[threadIdx.x] += st[threadIdx.x+s]; ss[threadIdx.x] += ss[threadIdx.x+s]; }
        __syncthreads();
    }
    if (threadIdx.x == 0) { cnts[combo] = st[0]; cnts[16 + combo] = ss[0]; }
}

// ---------------- kernel 1: bilinear 256->64 RGB + zero canvas ----------------
__global__ void k_prep(const float* __restrict__ t_rgb, const float* __restrict__ s_rgb,
                       float* __restrict__ tcol, float* __restrict__ scol,
                       float* __restrict__ canvas) {
    int idx = blockIdx.x * 256 + threadIdx.x;         // 0 .. 2*3*4096-1
    if (idx >= 2*3*HW) return;
    int i = idx & 4095;
    int c = (idx >> 12) % 3;
    int b = idx / (3*HW);
    int y0 = 4*(i >> 6) + 1, x0 = 4*(i & 63) + 1;     // half-pixel centers -> 2x2 avg
    const float* tp = t_rgb + (size_t)(b*3 + c) * 65536;
    const float* sp = s_rgb + (size_t)(b*3 + c) * 65536;
    tcol[idx] = 0.25f*(tp[y0*256+x0] + tp[y0*256+x0+1] + tp[(y0+1)*256+x0] + tp[(y0+1)*256+x0+1]);
    scol[idx] = 0.25f*(sp[y0*256+x0] + sp[y0*256+x0+1] + sp[(y0+1)*256+x0] + sp[(y0+1)*256+x0+1]);
    canvas[idx] = 0.f;
}

// ---------------- kernel 2: masked per-channel means ----------------
__global__ void k_means(const float* __restrict__ tfeat, const float* __restrict__ sfeat,
                        const float* __restrict__ tmaskA, const float* __restrict__ smaskA,
                        const float* __restrict__ cnts,
                        float* __restrict__ meanQ, float* __restrict__ meanS) {
    int c = blockIdx.x;                // channel
    int sc = blockIdx.y;               // 0..27
    int side = sc / NCOMBO, combo = sc % NCOMBO, b = combo / 7;
    bool use_self = cnts[16 + combo] < 9.f;
    const float* feat; const float* mask; float cnt;
    if (side == 0) {
        feat = tfeat + ((size_t)b*CDIM + c) * HW;
        mask = tmaskA + (size_t)combo*HW;
        cnt  = cnts[combo];
    } else {
        feat = (use_self ? tfeat : sfeat) + ((size_t)b*CDIM + c) * HW;
        mask = (use_self ? tmaskA : smaskA) + (size_t)combo*HW;
        cnt  = use_self ? cnts[combo] : cnts[16 + combo];
    }
    float acc = 0.f;
    for (int i = threadIdx.x; i < HW; i += 256) acc += feat[i] * mask[i];
    __shared__ float sm[256];
    sm[threadIdx.x] = acc; __syncthreads();
    for (int s = 128; s > 0; s >>= 1) {
        if (threadIdx.x < s) sm[threadIdx.x] += sm[threadIdx.x+s];
        __syncthreads();
    }
    if (threadIdx.x == 0)
        (side == 0 ? meanQ : meanS)[(size_t)combo*CDIM + c] = sm[0] / fmaxf(cnt, 1.f);
}

// ---------------- kernel 3: center, per-pixel L2 norm, emit row-major f16 ----------------
__global__ void k_norm(const float* __restrict__ tfeat, const float* __restrict__ sfeat,
                       const float* __restrict__ cnts,
                       const float* __restrict__ meanQ, const float* __restrict__ meanS,
                       _Float16* __restrict__ Qh, _Float16* __restrict__ Sh) {
    __shared__ float tile[32][257];    // 32 pixels x 256 channels, pad->stride 257 (odd)
    __shared__ float part[32][8];
    __shared__ float scl[32];
    int sc = blockIdx.y;
    int side = sc / NCOMBO, combo = sc % NCOMBO, b = combo / 7;
    bool use_self = cnts[16 + combo] < 9.f;
    const float* feat; const float* mean; float cnt;
    if (side == 0) {
        feat = tfeat + (size_t)b*CDIM*HW; mean = meanQ + (size_t)combo*CDIM; cnt = cnts[combo];
    } else {
        feat = (use_self ? tfeat : sfeat) + (size_t)b*CDIM*HW;
        mean = meanS + (size_t)combo*CDIM;
        cnt  = use_self ? cnts[combo] : cnts[16 + combo];
    }
    int base = blockIdx.x * 32;
    for (int idx = threadIdx.x; idx < 32*CDIM; idx += 256) {
        int c = idx >> 5, i = idx & 31;
        tile[i][c] = feat[(size_t)c*HW + base + i] - mean[c];
    }
    __syncthreads();
    {   // partial sum of squares: 8 groups of 32 channels per pixel
        int i = threadIdx.x & 31, g = threadIdx.x >> 5;
        float s = 0.f;
        for (int c = g*32; c < g*32 + 32; c++) { float v = tile[i][c]; s += v*v; }
        part[i][g] = s;
    }
    __syncthreads();
    if (threadIdx.x < 32) {
        float s = 0.f;
        for (int g = 0; g < 8; g++) s += part[threadIdx.x][g];
        float nrm = sqrtf(s);
        scl[threadIdx.x] = (cnt == 1.f) ? 1.f : 1.f / fmaxf(nrm, 1e-12f);
    }
    __syncthreads();
    _Float16* out = (side == 0 ? Qh : Sh) + (size_t)combo*HW*CDIM + (size_t)base*CDIM;
    for (int idx = threadIdx.x; idx < 32*CDIM; idx += 256) {
        int i = idx >> 8, c = idx & 255;
        out[(size_t)i*CDIM + c] = (_Float16)(tile[i][c] * scl[i]);
    }
}

// ---------------- kernel 4: fused WMMA attention + masked softmax + color mix ----------------
__global__ __launch_bounds__(32) void k_attn(
        const _Float16* __restrict__ Qh, const _Float16* __restrict__ Sh,
        const float* __restrict__ tmaskA, const float* __restrict__ smaskA,
        const float* __restrict__ cnts,
        const float* __restrict__ tcol, const float* __restrict__ scol,
        float* __restrict__ canvas) {
    int combo = blockIdx.y, b = combo / 7;
    int rowTile = blockIdx.x;
    int lane = threadIdx.x;
    int hi = lane >> 4;                 // 0: lanes 0-15, 1: lanes 16-31
    float t_cnt = cnts[combo];
    bool use_self = cnts[16 + combo] < 9.f;
    const float* tmask   = tmaskA + (size_t)combo*HW;
    const float* srcmask = use_self ? tmask : (smaskA + (size_t)combo*HW);
    const float* srccol  = (use_self ? tcol : scol) + (size_t)b*3*HW;
    const _Float16* Q = Qh + (size_t)combo*HW*CDIM;
    const _Float16* S = Sh + (size_t)combo*HW*CDIM;

    int row = rowTile*16 + (lane & 15);
    // Preload A fragments (16x32 f16 per K-step): lane<16 holds K {0-7,16-23}, lane>=16 {8-15,24-31}
    v16h a[8];
    {
        const _Float16* qr = Q + (size_t)row*CDIM + (hi ? 8 : 0);
        #pragma unroll
        for (int ks = 0; ks < 8; ks++) {
            *((v4f*)&a[ks])     = *(const v4f*)(qr + ks*32);
            *((v4f*)&a[ks] + 1) = *(const v4f*)(qr + ks*32 + 16);
        }
    }

    float l[8]  = {0,0,0,0,0,0,0,0};
    float ar[8] = {0,0,0,0,0,0,0,0};
    float ag[8] = {0,0,0,0,0,0,0,0};
    float ab[8] = {0,0,0,0,0,0,0,0};

    for (int jt = 0; jt < 256; jt++) {
        int col = jt*16 + (lane & 15);
        // B fragment (32x16 f16): lane<16 holds K {0-7,16-23} of its column, lane>=16 holds K {8-15,24-31}
        const _Float16* sr = S + (size_t)col*CDIM + (hi ? 8 : 0);
        v8f c = {0,0,0,0,0,0,0,0};
        #pragma unroll
        for (int ks = 0; ks < 8; ks++) {
            v16h bf;
            *((v4f*)&bf)     = *(const v4f*)(sr + ks*32);
            *((v4f*)&bf + 1) = *(const v4f*)(sr + ks*32 + 16);
            c = __builtin_amdgcn_wmma_f32_16x16x32_f16(false, a[ks], false, bf,
                                                       (short)0, c, false, false);
        }
        // scores are dot products of unit vectors (<=1): fixed-reference softmax, no max pass
        float mj = srcmask[col];
        float v0 = srccol[col], v1 = srccol[HW + col], v2 = srccol[2*HW + col];
        #pragma unroll
        for (int d = 0; d < 8; d++) {
            float e = mj * __expf(c[d]);       // masked -> exactly 0
            l[d]  += e;
            ar[d] += e * v0; ag[d] += e * v1; ab[d] += e * v2;
        }
    }

    // reduce across the 16 lanes of each half (xor 1,2,4,8 never crosses halves)
    #pragma unroll
    for (int d = 0; d < 8; d++) {
        #pragma unroll
        for (int m = 1; m < 16; m <<= 1) {
            l[d]  += __shfl_xor(l[d],  m, 32);
            ar[d] += __shfl_xor(ar[d], m, 32);
            ag[d] += __shfl_xor(ag[d], m, 32);
            ab[d] += __shfl_xor(ab[d], m, 32);
        }
    }
    // C layout: slot d = row d (half0) / row d+8 (half1); lane (lane&15)==d writes that row
    int s = lane & 15;
    if (s < 8 && t_cnt > 0.f) {
        int orow = rowTile*16 + s + (hi ? 8 : 0);
        if (tmask[orow] > 0.f) {                  // one-hot masks: disjoint across combos
            float inv = 1.f / l[s];
            float* cv = canvas + (size_t)b*3*HW;
            cv[orow]        = ar[s] * inv;
            cv[HW + orow]   = ag[s] * inv;
            cv[2*HW + orow] = ab[s] * inv;
        }
    }
}

// ---------------- kernel 5: bilinear 64->256 upsample + clip ----------------
__global__ void k_upsample(const float* __restrict__ canvas, float* __restrict__ out) {
    int idx = blockIdx.x * 256 + threadIdx.x;     // < 2*3*256*256
    if (idx >= 2*3*65536) return;
    int X = idx & 255, Y = (idx >> 8) & 255, bc = idx >> 16;
    const float* cv = canvas + (size_t)bc * HW;
    float sy = Y * 0.25f - 0.375f;
    float sx = X * 0.25f - 0.375f;
    int y0 = (int)floorf(sy); float fy = sy - (float)y0;
    int x0 = (int)floorf(sx); float fx = sx - (float)x0;
    int y0c = min(max(y0, 0), 63), y1c = min(max(y0 + 1, 0), 63);
    int x0c = min(max(x0, 0), 63), x1c = min(max(x0 + 1, 0), 63);
    float v00 = cv[y0c*64 + x0c], v01 = cv[y0c*64 + x1c];
    float v10 = cv[y1c*64 + x0c], v11 = cv[y1c*64 + x1c];
    float v = (1.f-fy)*((1.f-fx)*v00 + fx*v01) + fy*((1.f-fx)*v10 + fx*v11);
    out[idx] = fminf(1.f, fmaxf(-1.f, v));
}

// ---------------- launcher ----------------
extern "C" void kernel_launch(void* const* d_in, const int* in_sizes, int n_in,
                              void* d_out, int out_size, void* d_ws, size_t ws_size,
                              hipStream_t stream) {
    const float* s_rgb = (const float*)d_in[0];
    const float* t_rgb = (const float*)d_in[1];
    const float* s_oh  = (const float*)d_in[2];
    const float* t_oh  = (const float*)d_in[3];
    const float* tfeat = (const float*)d_in[4];
    const float* sfeat = (const float*)d_in[5];
    float* out = (float*)d_out;

    char* ws = (char*)d_ws;
    float*    tmask  = (float*)(ws + OFF_TMASK);
    float*    smask  = (float*)(ws + OFF_SMASK);
    float*    cnts   = (float*)(ws + OFF_CNTS);
    float*    tcol   = (float*)(ws + OFF_TCOL);
    float*    scol   = (float*)(ws + OFF_SCOL);
    float*    meanQ  = (float*)(ws + OFF_MEANQ);
    float*    meanS  = (float*)(ws + OFF_MEANS);
    _Float16* Qh     = (_Float16*)(ws + OFF_QH);
    _Float16* Sh     = (_Float16*)(ws + OFF_SH);
    float*    canvas = (float*)(ws + OFF_CANV);

    k_masks<<<NCOMBO, 256, 0, stream>>>(t_oh, s_oh, tmask, smask, cnts);
    k_prep <<<96,     256, 0, stream>>>(t_rgb, s_rgb, tcol, scol, canvas);
    dim3 gm(CDIM, 2*NCOMBO);
    k_means<<<gm, 256, 0, stream>>>(tfeat, sfeat, tmask, smask, cnts, meanQ, meanS);
    dim3 gn(HW/32, 2*NCOMBO);
    k_norm <<<gn, 256, 0, stream>>>(tfeat, sfeat, cnts, meanQ, meanS, Qh, Sh);
    dim3 ga(HW/16, NCOMBO);
    k_attn <<<ga, 32, 0, stream>>>(Qh, Sh, tmask, smask, cnts, tcol, scol, canvas);
    k_upsample<<<(2*3*65536)/256, 256, 0, stream>>>(canvas, out);
}
